// Attention_16707422781936
// MI455X (gfx1250) — matrix operand order repaired and checked by reading, not compiled, not run on hardware
//
#include <hip/hip_runtime.h>

typedef __attribute__((ext_vector_type(16))) _Float16 v16h;
typedef __attribute__((ext_vector_type(8)))  float    v8f;

union Frag16 {
    v16h v;
    unsigned u[8];
    _Float16 h[16];
};

__device__ __forceinline__ v8f wmma_f16(v16h a, v16h b, v8f c) {
    return __builtin_amdgcn_wmma_f32_16x16x32_f16(
        false, a, false, b, (short)0, c, false, false);
}

// ---------------------------------------------------------------------------
// TDM 2-D tile load (f16 elements) with vectorized-copy fallback.
// ---------------------------------------------------------------------------
#if defined(__has_builtin)
#  if __has_builtin(__builtin_amdgcn_tensor_load_to_lds)
#    define HAVE_TDM 1
#  endif
#  if __has_builtin(__builtin_amdgcn_s_wait_tensorcnt)
#    define HAVE_WTC 1
#  endif
#endif
#ifndef HAVE_TDM
#  define HAVE_TDM 0
#endif

__device__ __forceinline__ void wait_tensor0() {
#if defined(HAVE_WTC)
    __builtin_amdgcn_s_wait_tensorcnt(0);
#else
    asm volatile("s_wait_tensorcnt 0x0" ::: "memory");
#endif
}

#if HAVE_TDM
typedef unsigned u32x4 __attribute__((ext_vector_type(4)));
typedef int      i32x4 __attribute__((ext_vector_type(4)));
typedef int      i32x8 __attribute__((ext_vector_type(8)));

// Load a tile_h x tile_w tile of 2-byte elements (row stride = stride_elems)
// from gptr into LDS at byte offset lds_off.  D# per CDNA5 ISA ch.8.
__device__ __forceinline__ void tdm_load_2d_f16(
    unsigned lds_off, const _Float16* gptr,
    unsigned tile_w, unsigned tile_h, unsigned stride_elems)
{
    unsigned long long ga = (unsigned long long)(size_t)gptr;
    u32x4 g0;
    g0[0] = 1u;                                    // count=1, user descriptor
    g0[1] = lds_off;                               // lds_addr
    g0[2] = (unsigned)ga;                          // global_addr[31:0]
    g0[3] = (unsigned)((ga >> 32) & 0x01FFFFFFull) // global_addr[56:32]
            | (2u << 30);                          // type=2 ("image")
    i32x8 g1;
    g1[0] = (int)(1u << 16);                       // data_size=1 -> 2B elems
    g1[1] = 0;                                     // tensor_dim0[15:0] (big)
    g1[2] = (int)0x4000;                           // tensor_dim0 hi / dim1 lo
    g1[3] = (int)(0x4000u | (tile_w << 16));       // tensor_dim1 hi | tile_dim0
    g1[4] = (int)(tile_h & 0xFFFFu);               // tile_dim1 | tile_dim2=0
    g1[5] = (int)stride_elems;                     // tensor_dim0_stride lo32
    g1[6] = 0;
    g1[7] = 0;
    i32x4 gz  = {0, 0, 0, 0};
#if __clang_major__ >= 23
    i32x8 gz8 = {0, 0, 0, 0, 0, 0, 0, 0};
    __builtin_amdgcn_tensor_load_to_lds(g0, g1, gz, gz, gz8, 0);
#else
    __builtin_amdgcn_tensor_load_to_lds(g0, g1, gz, gz, 0);
#endif
}
#endif

// ---------------------------------------------------------------------------
// GEMM: Y[b][o][n] = sum_c W[o][c]*X[b][n][c] + bias[o]
// W [M,K] f16 row-major, X [B,N,K] f16 (n-major).  Out [B,M,N]: f32 or f16.
// 256 thr / 8 waves, 128(M) x 64(N) tile, K-step 32.
// TDM path: double-buffered LDS, DMA of step i+1 overlaps WMMAs of step i.
// ---------------------------------------------------------------------------
__global__ __launch_bounds__(256) void gemm_bias_wmma(
    const _Float16* __restrict__ Wm, const _Float16* __restrict__ X,
    const float* __restrict__ bias, float* __restrict__ Yf,
    _Float16* __restrict__ Yh, int M, int K, int N, int out_f16)
{
    __shared__ __align__(16) _Float16 lA[2][128 * 32];   // [m][k]
    __shared__ __align__(16) _Float16 lB[2][64 * 32];    // [n][k]

    const int t    = threadIdx.x;
    const int wid  = t >> 5;
    const int lane = t & 31;
    const int hi   = lane >> 4;
    const int l16  = lane & 15;

    const int bz = blockIdx.z;
    const int n0 = blockIdx.x * 64;
    const int m0 = blockIdx.y * 128;

    const _Float16* Xb = X + (size_t)bz * N * K;
    const _Float16* Wt = Wm + (size_t)m0 * K;
    const _Float16* Xt = Xb + (size_t)n0 * K;

    const v8f vzero = {0.f, 0.f, 0.f, 0.f, 0.f, 0.f, 0.f, 0.f};
    v8f acc[4] = {vzero, vzero, vzero, vzero};

    int p = 0;
#if HAVE_TDM
    if (wid == 0) {
        tdm_load_2d_f16((unsigned)(size_t)&lA[0][0], Wt, 32, 128, (unsigned)K);
        tdm_load_2d_f16((unsigned)(size_t)&lB[0][0], Xt, 32, 64, (unsigned)K);
        wait_tensor0();
    }
    __syncthreads();
#endif

    for (int kc = 0; kc < K; kc += 32) {
#if HAVE_TDM
        if (wid == 0 && kc + 32 < K) {          // prefetch next K-step
            tdm_load_2d_f16((unsigned)(size_t)&lA[p ^ 1][0], Wt + kc + 32,
                            32, 128, (unsigned)K);
            tdm_load_2d_f16((unsigned)(size_t)&lB[p ^ 1][0], Xt + kc + 32,
                            32, 64, (unsigned)K);
        }
        const _Float16* cA = &lA[p][0];
        const _Float16* cB = &lB[p][0];
#else
        __syncthreads();
        {
            uint4* dA = (uint4*)&lA[0][0];      // 512 x uint4
            #pragma unroll
            for (int e = 0; e < 2; ++e) {
                int f = e * 256 + t;
                int m = f >> 2, seg = f & 3;
                dA[f] = *(const uint4*)(Wt + (size_t)m * K + kc + seg * 8);
            }
            uint4* dB = (uint4*)&lB[0][0];      // 256 x uint4
            {
                int f = t;
                int n = f >> 2, seg = f & 3;
                dB[f] = *(const uint4*)(Xt + (size_t)n * K + kc + seg * 8);
            }
        }
        __syncthreads();
        const _Float16* cA = &lA[0][0];
        const _Float16* cB = &lB[0][0];
#endif
        // A fragment: rows m0 + wid*16 + l16  (2x ds_load_b128)
        Frag16 af;
        {
            const unsigned* pa = (const unsigned*)&cA[(wid * 16 + l16) * 32];
            #pragma unroll
            for (int i = 0; i < 4; ++i) af.u[i]     = pa[hi * 8 + i];
            #pragma unroll
            for (int i = 0; i < 4; ++i) af.u[4 + i] = pa[hi * 8 + 4 + i];
        }
        #pragma unroll
        for (int j = 0; j < 4; ++j) {
            // B fragment: col n = j*16 + l16  (2x ds_load_b128)
            Frag16 bf;
            const unsigned* pb = (const unsigned*)&cB[(j * 16 + l16) * 32];
            #pragma unroll
            for (int i = 0; i < 8; ++i) bf.u[i] = pb[hi * 8 + i];
            acc[j] = wmma_f16(af.v, bf.v, acc[j]);
        }
#if HAVE_TDM
        if (wid == 0) wait_tensor0();           // next tiles landed
        __syncthreads();                        // + everyone done reading p
        p ^= 1;
#endif
    }

    const int obase = m0 + wid * 16 + hi * 8;
    if (out_f16) {
        _Float16* Yb = Yh + (size_t)bz * M * N;
        #pragma unroll
        for (int j = 0; j < 4; ++j) {
            int n = n0 + j * 16 + l16;
            #pragma unroll
            for (int r = 0; r < 8; ++r) {
                int o = obase + r;
                Yb[(size_t)o * N + n] = (_Float16)(acc[j][r] + bias[o]);
            }
        }
    } else {
        float* Yb = Yf + (size_t)bz * M * N;
        #pragma unroll
        for (int j = 0; j < 4; ++j) {
            int n = n0 + j * 16 + l16;
            #pragma unroll
            for (int r = 0; r < 8; ++r) {
                int o = obase + r;
                Yb[(size_t)o * N + n] = acc[j][r] + bias[o];
            }
        }
    }
}

// ---------------------------------------------------------------------------
// Flash attention, f16 in, f16 out in [B][N][512] (n-major) layout.
// qkv_h [B,1024,N]: q rows h*32, k rows 256+h*32, v rows 512+h*64.
// WG = 4 waves = 4 adjacent 16-query tiles of one (b,h).
// TDM path: double-buffered K/V chunk prefetch overlaps S/softmax/PV.
// ---------------------------------------------------------------------------
__global__ __launch_bounds__(128) void flash_attn_wmma(
    const _Float16* __restrict__ qkv, _Float16* __restrict__ outv, int N)
{
    __shared__ __align__(16) _Float16 lK[2][32 * 32];   // [kd][m] 2x2KB
    __shared__ __align__(16) _Float16 lV[2][64 * 32];   // [d][m]  2x4KB
    __shared__ __align__(16) _Float16 lP[4][16 * 32];   // per-wave P tile

    const int t = threadIdx.x, wid = t >> 5, lane = t & 31;
    const int hi = lane >> 4, l16 = lane & 15;

    int gb = blockIdx.x;
    const int ngroups = N >> 6;                 // 25 groups of 4 n-tiles
    const int ng = gb % ngroups;  gb /= ngroups;
    const int h  = gb & 7;
    const int b  = gb >> 3;
    const int nt = ng * 4 + wid;

    const _Float16* qb = qkv + ((size_t)b * 1024 + h * 32) * N;
    const _Float16* kb = qkv + ((size_t)b * 1024 + 256 + h * 32) * N;
    const _Float16* vb = qkv + ((size_t)b * 1024 + 512 + h * 64) * N;

    const float sscale = 0.17677669529663687f;  // 1/sqrt(32)

    // Q as A fragment (16 queries x kd=32)
    Frag16 qa;
    {
        const int n = nt * 16 + l16;
        #pragma unroll
        for (int i = 0; i < 8; ++i) {
            const int k0 = (i < 4) ? (hi * 8 + 2 * i) : (16 + hi * 8 + 2 * (i - 4));
            union { unsigned u; _Float16 hh[2]; } p;
            p.hh[0] = qb[(size_t)k0 * N + n];
            p.hh[1] = qb[(size_t)(k0 + 1) * N + n];
            qa.u[i] = p.u;
        }
    }

    const v8f vzero = {0.f, 0.f, 0.f, 0.f, 0.f, 0.f, 0.f, 0.f};
    v8f oacc[4] = {vzero, vzero, vzero, vzero};
    float mrun[8], lrun[8];
    #pragma unroll
    for (int r = 0; r < 8; ++r) { mrun[r] = -1e30f; lrun[r] = 0.f; }

    _Float16* lPw = &lP[wid][0];

    int p = 0;
#if HAVE_TDM
    if (wid == 0) {
        tdm_load_2d_f16((unsigned)(size_t)&lK[0][0], kb, 32, 32, (unsigned)N);
        tdm_load_2d_f16((unsigned)(size_t)&lV[0][0], vb, 32, 64, (unsigned)N);
        wait_tensor0();
    }
    __syncthreads();
#endif

    for (int mc = 0; mc < N; mc += 32) {
#if HAVE_TDM
        if (wid == 0 && mc + 32 < N) {          // prefetch next key chunk
            tdm_load_2d_f16((unsigned)(size_t)&lK[p ^ 1][0], kb + mc + 32,
                            32, 32, (unsigned)N);
            tdm_load_2d_f16((unsigned)(size_t)&lV[p ^ 1][0], vb + mc + 32,
                            32, 64, (unsigned)N);
        }
        const _Float16* cK = &lK[p][0];
        const _Float16* cV = &lV[p][0];
#else
        __syncthreads();
        {
            uint4* dK = (uint4*)&lK[0][0];      // 128 x uint4
            int f = t;
            dK[f] = *(const uint4*)(kb + (size_t)(f >> 2) * N + mc + (f & 3) * 8);
            uint4* dV = (uint4*)&lV[0][0];      // 256 x uint4
            #pragma unroll
            for (int e = 0; e < 2; ++e) {
                int g = e * 128 + t;
                dV[g] = *(const uint4*)(vb + (size_t)(g >> 2) * N + mc + (g & 3) * 8);
            }
        }
        __syncthreads();
        const _Float16* cK = &lK[0][0];
        const _Float16* cV = &lV[0][0];
#endif
        // K fragments for key sub-tiles [0,16) and [16,32)
        Frag16 bk0, bk1;
        #pragma unroll
        for (int i = 0; i < 8; ++i) {
            const int kd = (i < 4) ? (hi * 8 + 2 * i) : (16 + hi * 8 + 2 * (i - 4));
            union { unsigned u; _Float16 hh[2]; } p0, p1;
            p0.hh[0] = cK[kd * 32 + l16];
            p0.hh[1] = cK[(kd + 1) * 32 + l16];
            p1.hh[0] = cK[kd * 32 + 16 + l16];
            p1.hh[1] = cK[(kd + 1) * 32 + 16 + l16];
            bk0.u[i] = p0.u;  bk1.u[i] = p1.u;
        }

        v8f s0 = wmma_f16(qa.v, bk0.v, vzero);
        v8f s1 = wmma_f16(qa.v, bk1.v, vzero);

        // Online softmax; rows live per 16-lane half -> xor-1/2/4/8 tree
        #pragma unroll
        for (int r = 0; r < 8; ++r) {
            const float a0 = s0[r] * sscale;
            const float a1 = s1[r] * sscale;
            float mx = fmaxf(a0, a1);
            mx = fmaxf(mx, __shfl_xor(mx, 1, 32));
            mx = fmaxf(mx, __shfl_xor(mx, 2, 32));
            mx = fmaxf(mx, __shfl_xor(mx, 4, 32));
            mx = fmaxf(mx, __shfl_xor(mx, 8, 32));
            const float mn    = fmaxf(mrun[r], mx);
            const float alpha = __expf(mrun[r] - mn);
            mrun[r] = mn;
            const float p0 = __expf(a0 - mn);
            const float p1 = __expf(a1 - mn);
            float rs = p0 + p1;
            rs += __shfl_xor(rs, 1, 32);
            rs += __shfl_xor(rs, 2, 32);
            rs += __shfl_xor(rs, 4, 32);
            rs += __shfl_xor(rs, 8, 32);
            lrun[r] = lrun[r] * alpha + rs;
            oacc[0][r] *= alpha;  oacc[1][r] *= alpha;
            oacc[2][r] *= alpha;  oacc[3][r] *= alpha;
            const int nrow = r + hi * 8;
            lPw[nrow * 32 + l16]      = (_Float16)p0;
            lPw[nrow * 32 + 16 + l16] = (_Float16)p1;
        }

        asm volatile("s_wait_dscnt 0" ::: "memory");

        // Reload P as A fragment
        Frag16 pf;
        {
            const unsigned* pp = (const unsigned*)&lPw[l16 * 32];
            #pragma unroll
            for (int i = 0; i < 4; ++i) pf.u[i]     = pp[hi * 8 + i];
            #pragma unroll
            for (int i = 0; i < 4; ++i) pf.u[4 + i] = pp[hi * 8 + 4 + i];
        }

        // O += P @ V^T : 4 d-tiles, K = 32 keys from LDS
        #pragma unroll
        for (int dt = 0; dt < 4; ++dt) {
            Frag16 vf;
            const _Float16* vrow = &cV[(dt * 16 + l16) * 32 + hi * 16];
            #pragma unroll
            for (int i = 0; i < 8; ++i)
                vf.u[i] = *(const unsigned*)(vrow + 2 * i);
            oacc[dt] = wmma_f16(pf.v, vf.v, oacc[dt]);
        }

#if HAVE_TDM
        if (wid == 0) wait_tensor0();
        __syncthreads();
        p ^= 1;
#endif
    }

    // Store n-major: attnh[b][n][h*64+d]  (lane-contiguous along d)
    _Float16* ob = outv + (size_t)b * N * 512 + h * 64;
    #pragma unroll
    for (int dt = 0; dt < 4; ++dt) {
        const int d = dt * 16 + l16;
        #pragma unroll
        for (int r = 0; r < 8; ++r) {
            const int n = nt * 16 + r + hi * 8;
            ob[(size_t)n * 512 + d] = (_Float16)(oacc[dt][r] / lrun[r]);
        }
    }
}

// ---------------------------------------------------------------------------
// Transpose + convert: src [B][C][N] f32 -> dst [B][N][C] f16, 32x32 tiles
// ---------------------------------------------------------------------------
__global__ __launch_bounds__(256) void transpose_cvt_f32_f16(
    const float* __restrict__ src, _Float16* __restrict__ dst, int C, int N)
{
    __shared__ float lT[32][33];
    const int t  = threadIdx.x;
    const int tr = t >> 5;          // 0..7
    const int tc = t & 31;
    const int n0 = blockIdx.x * 32;
    const int c0 = blockIdx.y * 32;
    const int b  = blockIdx.z;
    const float* sb = src + ((size_t)b * C + c0) * N + n0;
    _Float16*    db = dst + ((size_t)b * N + n0) * C + c0;
    #pragma unroll
    for (int it = 0; it < 4; ++it) {
        int c = it * 8 + tr;
        lT[tc][c] = sb[(size_t)c * N + tc];     // coalesced read along n
    }
    __syncthreads();
    #pragma unroll
    for (int it = 0; it < 4; ++it) {
        int n = it * 8 + tr;
        db[(size_t)n * C + tc] = (_Float16)lT[n][tc];   // coalesced write along c
    }
}

// ---------------------------------------------------------------------------
// Plain f32 -> f16 conversion for weights, 4 elements/thread
// ---------------------------------------------------------------------------
__global__ void cvt_f32_f16(const float* __restrict__ s,
                            _Float16* __restrict__ d, int n)
{
    int i = (blockIdx.x * blockDim.x + threadIdx.x) * 4;
    if (i >= n) return;
    float4 v = *(const float4*)(s + i);
    union { _Float16 h[4]; unsigned long long u; } p;
    p.h[0] = (_Float16)v.x;  p.h[1] = (_Float16)v.y;
    p.h[2] = (_Float16)v.z;  p.h[3] = (_Float16)v.w;
    *(unsigned long long*)(d + i) = p.u;
}

// ---------------------------------------------------------------------------
// Depthwise 3x3 PE conv on v (f16 rows 512..1023 of qkv_h), += into d_out
// ---------------------------------------------------------------------------
__global__ void dwconv_pe_add(const _Float16* __restrict__ qkvh,
                              const float* __restrict__ Wpe,
                              const float* __restrict__ bpe,
                              float* __restrict__ outp,
                              int H, int W, int N)
{
    int idx = blockIdx.x * blockDim.x + threadIdx.x;
    const int total = 8 * 512 * N;
    if (idx >= total) return;
    const int n = idx % N;
    const int c = (idx / N) & 511;
    const int b = idx / (N * 512);
    const int x = n % W, y = n / W;
    const _Float16* src = qkvh + ((size_t)b * 1024 + 512 + c) * N;
    const float* wp = Wpe + c * 9;
    float s = bpe[c];
    #pragma unroll
    for (int dy = -1; dy <= 1; ++dy) {
        const int yy = y + dy;
        if (yy < 0 || yy >= H) continue;
        #pragma unroll
        for (int dx = -1; dx <= 1; ++dx) {
            const int xx = x + dx;
            if (xx < 0 || xx >= W) continue;
            s += (float)src[yy * W + xx] * wp[(dy + 1) * 3 + (dx + 1)];
        }
    }
    outp[idx] += s;
}

// ---------------------------------------------------------------------------
extern "C" void kernel_launch(void* const* d_in, const int* in_sizes, int n_in,
                              void* d_out, int out_size, void* d_ws, size_t ws_size,
                              hipStream_t stream)
{
    (void)in_sizes; (void)n_in; (void)out_size; (void)ws_size;

    const float* x     = (const float*)d_in[0];
    const float* Wqkv  = (const float*)d_in[1];
    const float* bqkv  = (const float*)d_in[2];
    const float* Wproj = (const float*)d_in[3];
    const float* bproj = (const float*)d_in[4];
    const float* Wpe   = (const float*)d_in[5];
    const float* bpe   = (const float*)d_in[6];
    float* out = (float*)d_out;

    const int B = 8, C = 512, H = 40, W = 40;
    const int N  = H * W;        // 1600
    const int Hq = 1024;

    // f16 workspace layout (all 16B-aligned)
    _Float16* xh    = (_Float16*)d_ws;                       // [B][N][C]
    _Float16* Wqh   = xh   + (size_t)B * C * N;              // [1024][512]
    _Float16* Wph   = Wqh  + (size_t)Hq * C;                 // [512][512]
    _Float16* qkvh  = Wph  + (size_t)C * C;                  // [B][1024][N]
    _Float16* attnh = qkvh + (size_t)B * Hq * N;             // [B][N][512]

    // x: transpose to n-major + convert
    transpose_cvt_f32_f16<<<dim3(N / 32, C / 32, B), 256, 0, stream>>>(x, xh, C, N);
    cvt_f32_f16<<<(Hq * C / 4 + 255) / 256, 256, 0, stream>>>(Wqkv, Wqh, Hq * C);
    cvt_f32_f16<<<(C * C / 4 + 255) / 256, 256, 0, stream>>>(Wproj, Wph, C * C);

    // 1) QKV projection -> f16 [B][1024][N]
    gemm_bias_wmma<<<dim3(N / 64, Hq / 128, B), 256, 0, stream>>>(
        Wqh, xh, bqkv, nullptr, qkvh, Hq, C, N, 1);

    // 2) Flash attention -> f16 [B][N][512]
    flash_attn_wmma<<<dim3(B * 8 * (N / 64)), 128, 0, stream>>>(qkvh, attnh, N);

    // 3) Output projection -> f32 d_out [B][512][N]
    gemm_bias_wmma<<<dim3(N / 64, C / 128, B), 256, 0, stream>>>(
        Wph, attnh, bproj, out, nullptr, C, C, N, 0);

    // 4) Depthwise 3x3 PE conv, accumulate into d_out
    const int total = B * C * N;
    dwconv_pe_add<<<(total + 255) / 256, 256, 0, stream>>>(
        qkvh, Wpe, bpe, out, H, W, N);
}
